// GPT2PagedModel_575525617895
// MI455X (gfx1250) — matrix operand order repaired and checked by reading, not compiled, not run on hardware
//
#include <hip/hip_runtime.h>
#include <hip/hip_bf16.h>
#include <math.h>

// ---------------- types ----------------
typedef __bf16 bf16;
typedef __attribute__((ext_vector_type(16))) __bf16 v16bf;
typedef __attribute__((ext_vector_type(8)))  float  v8f;

#define NL 12
#define NH 12
#define HD 64
#define DMODEL 768
#define SEQ 1024
#define VOCAB 50257
#define VPAD 50272
#define SCALE 0.125f /* 1/sqrt(64) */

// fp32 -> bf16 round-to-nearest-even
__device__ __forceinline__ bf16 f2bf(float f) {
  union { float f; unsigned u; } x; x.f = f;
  unsigned r = x.u + 0x7FFFu + ((x.u >> 16) & 1u);
  union { unsigned short s; bf16 b; } y; y.s = (unsigned short)(r >> 16);
  return y.b;
}

__device__ __forceinline__ float gelu_tanh(float x) {
  float x3 = x * x * x;
  return 0.5f * x * (1.f + tanhf(0.7978845608028654f * (x + 0.044715f * x3)));
}

union Frag { v16bf v; uint4 q[2]; };

// A fragment (16x32 bf16): row = lane&15, K chunks at (lane>>4)*8 and +16
__device__ __forceinline__ v16bf load_a(const bf16* __restrict__ p0, int lda) {
  int lane = threadIdx.x & 31;
  const bf16* p = p0 + (size_t)(lane & 15) * lda + ((lane >> 4) * 8);
  Frag f;
  f.q[0] = *(const uint4*)(p);
  f.q[1] = *(const uint4*)(p + 16);
  return f.v;
}

// B fragment (32x16 bf16) from WT[n][k]: col = lane&15, K chunks at (lane>>4)*8 and +16
__device__ __forceinline__ v16bf load_b(const bf16* __restrict__ p0, int ldb) {
  int lane = threadIdx.x & 31;
  const bf16* p = p0 + (size_t)(lane & 15) * ldb + ((lane >> 4) * 8);
  Frag f;
  f.q[0] = *(const uint4*)(p);
  f.q[1] = *(const uint4*)(p + 16);
  return f.v;
}

// MT 16-row tiles of A vs NT 16-col tiles of B.
// Ping-pong double buffered, k-loop unrolled by 2 (requires K % 64 == 0):
// no inter-buffer register copies, loads for the next 32-chunk are in flight
// while the current chunk's WMMAs issue.
template <int MT, int NT>
__device__ __forceinline__ void gemm_loop(const bf16* __restrict__ Abase, int lda,
                                          const bf16* __restrict__ Bbase, int ldb,
                                          int K, v8f* acc /* [MT*NT] */) {
  v16bf a0[MT], b0[NT], a1[MT], b1[NT];
#pragma unroll
  for (int j = 0; j < NT; ++j) b0[j] = load_b(Bbase + (size_t)(16 * j) * ldb, ldb);
#pragma unroll
  for (int t = 0; t < MT; ++t) a0[t] = load_a(Abase + (size_t)(16 * t) * lda, lda);
  int k = 0;
  for (; k + 64 < K; k += 64) {
#pragma unroll
    for (int j = 0; j < NT; ++j)
      b1[j] = load_b(Bbase + (size_t)(16 * j) * ldb + k + 32, ldb);
#pragma unroll
    for (int t = 0; t < MT; ++t)
      a1[t] = load_a(Abase + (size_t)(16 * t) * lda + k + 32, lda);
#pragma unroll
    for (int t = 0; t < MT; ++t)
#pragma unroll
      for (int j = 0; j < NT; ++j)
        acc[t * NT + j] = __builtin_amdgcn_wmma_f32_16x16x32_bf16(
            false, a0[t], false, b0[j], (short)0, acc[t * NT + j], false, false);
#pragma unroll
    for (int j = 0; j < NT; ++j)
      b0[j] = load_b(Bbase + (size_t)(16 * j) * ldb + k + 64, ldb);
#pragma unroll
    for (int t = 0; t < MT; ++t)
      a0[t] = load_a(Abase + (size_t)(16 * t) * lda + k + 64, lda);
#pragma unroll
    for (int t = 0; t < MT; ++t)
#pragma unroll
      for (int j = 0; j < NT; ++j)
        acc[t * NT + j] = __builtin_amdgcn_wmma_f32_16x16x32_bf16(
            false, a1[t], false, b1[j], (short)0, acc[t * NT + j], false, false);
  }
  // tail: exactly 64 K-values remain
#pragma unroll
  for (int j = 0; j < NT; ++j)
    b1[j] = load_b(Bbase + (size_t)(16 * j) * ldb + k + 32, ldb);
#pragma unroll
  for (int t = 0; t < MT; ++t)
    a1[t] = load_a(Abase + (size_t)(16 * t) * lda + k + 32, lda);
#pragma unroll
  for (int t = 0; t < MT; ++t)
#pragma unroll
    for (int j = 0; j < NT; ++j)
      acc[t * NT + j] = __builtin_amdgcn_wmma_f32_16x16x32_bf16(
          false, a0[t], false, b0[j], (short)0, acc[t * NT + j], false, false);
#pragma unroll
  for (int t = 0; t < MT; ++t)
#pragma unroll
    for (int j = 0; j < NT; ++j)
      acc[t * NT + j] = __builtin_amdgcn_wmma_f32_16x16x32_bf16(
          false, a1[t], false, b1[j], (short)0, acc[t * NT + j], false, false);
}

// ---------------- elementwise / prep kernels ----------------
__global__ void k_embed(const int* __restrict__ ids, const float* __restrict__ wte,
                        const float* __restrict__ wpe, float* __restrict__ x) {
  int i = blockIdx.x * blockDim.x + threadIdx.x;
  if (i >= SEQ * DMODEL) return;
  int l = i / DMODEL, d = i % DMODEL;
  x[i] = wte[(size_t)ids[l] * DMODEL + d] + wpe[i];
}

// WT[n*K + k] = bf16(W[k*N + n])  (transpose-convert)
__global__ void k_convT(const float* __restrict__ W, bf16* __restrict__ WT, int K, int N) {
  int i = blockIdx.x * blockDim.x + threadIdx.x;
  if (i >= N * K) return;
  int n = i / K, k = i % K;
  WT[i] = f2bf(W[(size_t)k * N + n]);
}

// wteT[r*K + k] = bf16(wte[r*K + k]) for r < valid else 0 (zero-pad rows)
__global__ void k_convPad(const float* __restrict__ W, bf16* __restrict__ WT,
                          size_t total, size_t validElems) {
  size_t i = (size_t)blockIdx.x * blockDim.x + threadIdx.x;
  if (i >= total) return;
  WT[i] = (i < validElems) ? f2bf(W[i]) : f2bf(0.f);
}

__global__ void k_layernorm(const float* __restrict__ x, const float* __restrict__ g,
                            const float* __restrict__ b, bf16* __restrict__ out) {
  __shared__ float red[256];
  int row = blockIdx.x, tid = threadIdx.x;
  const float* xr = x + (size_t)row * DMODEL;
  float v0 = xr[tid], v1 = xr[tid + 256], v2 = xr[tid + 512];
  red[tid] = v0 + v1 + v2;
  __syncthreads();
  for (int st = 128; st > 0; st >>= 1) {
    if (tid < st) red[tid] += red[tid + st];
    __syncthreads();
  }
  float mean = red[0] / (float)DMODEL;
  __syncthreads();
  red[tid] = v0 * v0 + v1 * v1 + v2 * v2;
  __syncthreads();
  for (int st = 128; st > 0; st >>= 1) {
    if (tid < st) red[tid] += red[tid + st];
    __syncthreads();
  }
  float var = red[0] / (float)DMODEL - mean * mean;
  float inv = rsqrtf(var + 1e-5f);
  bf16* o = out + (size_t)row * DMODEL;
  o[tid]       = f2bf((v0 - mean) * inv * g[tid]       + b[tid]);
  o[tid + 256] = f2bf((v1 - mean) * inv * g[tid + 256] + b[tid + 256]);
  o[tid + 512] = f2bf((v2 - mean) * inv * g[tid + 512] + b[tid + 512]);
}

// ---------------- GEMM kernels (WMMA) ----------------
// QKV: [1024x768] @ WT[2304x768] + bias, scatter to Qb/Kb/VT(bf16)
// wave computes 64 rows x 32 cols (4x2 tiles)
__global__ __launch_bounds__(128) void k_gemm_qkv(
    const bf16* __restrict__ A, const bf16* __restrict__ WT,
    const float* __restrict__ bias, bf16* __restrict__ qb,
    bf16* __restrict__ kb, bf16* __restrict__ vtb) {
  int wave = threadIdx.x >> 5;
  int n0 = (blockIdx.x * 4 + wave) * 32;
  int m0 = blockIdx.y * 64;
  v8f acc[8] = {};
  gemm_loop<4, 2>(A + (size_t)m0 * DMODEL, DMODEL,
                  WT + (size_t)n0 * DMODEL, DMODEL, DMODEL, acc);
  int lane = threadIdx.x & 31;
#pragma unroll
  for (int j = 0; j < 2; ++j) {
    int n = n0 + j * 16 + (lane & 15);
    float bv = bias[n];
#pragma unroll
    for (int t = 0; t < 4; ++t)
#pragma unroll
      for (int r = 0; r < 8; ++r) {
        int m = m0 + t * 16 + r + ((lane >> 4) << 3);
        float v = acc[t * 2 + j][r] + bv;
        if (n < DMODEL)          qb[(size_t)m * DMODEL + n] = f2bf(v);
        else if (n < 2 * DMODEL) kb[(size_t)m * DMODEL + (n - DMODEL)] = f2bf(v);
        else { int c = n - 2 * DMODEL; vtb[(size_t)c * SEQ + m] = f2bf(v); }
      }
  }
}

// MODE 0: outF = resid + A@WT + bias
// MODE 1: outB = bf16(gelu(A@WT + bias))
// MODE 2: outF = A@WT (n < Nvalid)
template <int MODE>
__global__ __launch_bounds__(128) void k_gemm(
    const bf16* __restrict__ A, const bf16* __restrict__ WT,
    const float* __restrict__ bias, const float* __restrict__ resid,
    float* __restrict__ outF, bf16* __restrict__ outB,
    int N, int K, int ldc, int Nvalid) {
  int wave = threadIdx.x >> 5;
  int n0 = (blockIdx.x * 4 + wave) * 32;
  if (n0 >= N) return;
  int m0 = blockIdx.y * 64;
  v8f acc[8] = {};
  gemm_loop<4, 2>(A + (size_t)m0 * K, K, WT + (size_t)n0 * K, K, K, acc);
  int lane = threadIdx.x & 31;
#pragma unroll
  for (int j = 0; j < 2; ++j) {
    int n = n0 + j * 16 + (lane & 15);
    float bv = (MODE == 2) ? 0.f : bias[n];
#pragma unroll
    for (int t = 0; t < 4; ++t)
#pragma unroll
      for (int r = 0; r < 8; ++r) {
        int m = m0 + t * 16 + r + ((lane >> 4) << 3);
        float v = acc[t * 2 + j][r] + bv;
        if (MODE == 0) outF[(size_t)m * ldc + n] = resid[(size_t)m * ldc + n] + v;
        if (MODE == 1) outB[(size_t)m * ldc + n] = f2bf(gelu_tanh(v));
        if (MODE == 2) { if (n < Nvalid) outF[(size_t)m * ldc + n] = v; }
      }
  }
}

// scores[h][m][t] = SCALE * q[m,h,:].k[t,h,:], causal mask.
// one wave per 64-row slab x 16-col tile
__global__ __launch_bounds__(32) void k_scores(const bf16* __restrict__ qb,
                                               const bf16* __restrict__ kb,
                                               float* __restrict__ scores) {
  int t0 = blockIdx.x * 16, m0 = blockIdx.y * 64, h = blockIdx.z;
  if (t0 > m0 + 63) return;  // fully masked slab
  v8f acc[4] = {};
  gemm_loop<4, 1>(qb + (size_t)m0 * DMODEL + h * HD, DMODEL,
                  kb + (size_t)t0 * DMODEL + h * HD, DMODEL, HD, acc);
  int lane = threadIdx.x & 31;
  int n = t0 + (lane & 15);
  float* srow = scores + (size_t)h * SEQ * SEQ;
#pragma unroll
  for (int t = 0; t < 4; ++t)
#pragma unroll
    for (int r = 0; r < 8; ++r) {
      int m = m0 + t * 16 + r + ((lane >> 4) << 3);
      float s = acc[t][r] * SCALE;
      if (n > m) s = -3.0e38f;
      srow[(size_t)m * SEQ + n] = s;
    }
}

// per-row softmax over t<=l; writes bf16 P zero-padded to T
__global__ void k_softmax(const float* __restrict__ scores, bf16* __restrict__ p) {
  __shared__ float red[256];
  int row = blockIdx.x;          // h*SEQ + l
  int l = row & (SEQ - 1);
  int nvalid = l + 1;
  int tid = threadIdx.x;
  const float* s = scores + (size_t)row * SEQ;
  bf16* pr = p + (size_t)row * SEQ;
  float mx = -3.4e38f;
  for (int t = tid; t < nvalid; t += 256) mx = fmaxf(mx, s[t]);
  red[tid] = mx; __syncthreads();
  for (int st = 128; st > 0; st >>= 1) {
    if (tid < st) red[tid] = fmaxf(red[tid], red[tid + st]);
    __syncthreads();
  }
  mx = red[0]; __syncthreads();
  float sum = 0.f;
  for (int t = tid; t < nvalid; t += 256) sum += __expf(s[t] - mx);
  red[tid] = sum; __syncthreads();
  for (int st = 128; st > 0; st >>= 1) {
    if (tid < st) red[tid] += red[tid + st];
    __syncthreads();
  }
  float inv = 1.f / red[0];
  for (int t = tid; t < SEQ; t += 256)
    pr[t] = f2bf(t < nvalid ? __expf(s[t] - mx) * inv : 0.f);
}

// out[m, h*64+d] = sum_t P[h][m][t] * VT[h][d][t]; causal K-loop bound
// wave computes 64 rows x 32 d-cols (4x2 tiles)
__global__ __launch_bounds__(32) void k_attnout(const bf16* __restrict__ p,
                                                const bf16* __restrict__ vtb,
                                                bf16* __restrict__ ao) {
  int n0 = blockIdx.x * 32;      // d pair-tile (0..1)
  int m0 = blockIdx.y * 64;      // 64-row slab
  int h  = blockIdx.z;
  v8f acc[8] = {};
  const bf16* A = p   + (size_t)h * SEQ * SEQ;
  const bf16* B = vtb + (size_t)h * HD * SEQ;
  gemm_loop<4, 2>(A + (size_t)m0 * SEQ, SEQ, B + (size_t)n0 * SEQ, SEQ,
                  m0 + 64 /* P is zero beyond row's length */, acc);
  int lane = threadIdx.x & 31;
#pragma unroll
  for (int j = 0; j < 2; ++j) {
    int n = n0 + j * 16 + (lane & 15);
#pragma unroll
    for (int t = 0; t < 4; ++t)
#pragma unroll
      for (int r = 0; r < 8; ++r) {
        int m = m0 + t * 16 + r + ((lane >> 4) << 3);
        ao[(size_t)m * DMODEL + h * HD + n] = f2bf(acc[t * 2 + j][r]);
      }
  }
}

// ---------------- host orchestration ----------------
extern "C" void kernel_launch(void* const* d_in, const int* in_sizes, int n_in,
                              void* d_out, int out_size, void* d_ws, size_t ws_size,
                              hipStream_t stream) {
  const int*   ids   = (const int*)d_in[0];
  const float* wte   = (const float*)d_in[5];
  const float* wpe   = (const float*)d_in[6];
  const float* ln1g  = (const float*)d_in[7];
  const float* ln1b  = (const float*)d_in[8];
  const float* attw  = (const float*)d_in[9];
  const float* attb  = (const float*)d_in[10];
  const float* pw    = (const float*)d_in[11];
  const float* pb    = (const float*)d_in[12];
  const float* ln2g  = (const float*)d_in[13];
  const float* ln2b  = (const float*)d_in[14];
  const float* fcw   = (const float*)d_in[15];
  const float* fcb   = (const float*)d_in[16];
  const float* prw   = (const float*)d_in[17];
  const float* prb   = (const float*)d_in[18];
  const float* lnfg  = (const float*)d_in[19];
  const float* lnfb  = (const float*)d_in[20];
  float* logits = (float*)d_out;

  char* ws = (char*)d_ws;
  size_t off = 0;
  auto take = [&](size_t bytes) -> char* {
    char* q = ws + off;
    off = (off + bytes + 255) & ~(size_t)255;
    return q;
  };
  float* x    = (float*)take((size_t)SEQ * DMODEL * 4);
  bf16*  h    = (bf16*) take((size_t)SEQ * DMODEL * 2);
  bf16*  qb   = (bf16*) take((size_t)SEQ * DMODEL * 2);
  bf16*  kb   = (bf16*) take((size_t)SEQ * DMODEL * 2);
  bf16*  vtb  = (bf16*) take((size_t)DMODEL * SEQ * 2);
  bf16*  ao   = (bf16*) take((size_t)SEQ * DMODEL * 2);
  bf16*  a4   = (bf16*) take((size_t)SEQ * 4 * DMODEL * 2);
  float* sc   = (float*)take((size_t)NH * SEQ * SEQ * 4);
  bf16*  pbuf = (bf16*) take((size_t)NH * SEQ * SEQ * 2);
  bf16*  wtq  = (bf16*) take((size_t)3 * DMODEL * DMODEL * 2);
  bf16*  wtp  = (bf16*) take((size_t)DMODEL * DMODEL * 2);
  bf16*  wtf  = (bf16*) take((size_t)4 * DMODEL * DMODEL * 2);
  bf16*  wtr  = (bf16*) take((size_t)DMODEL * 4 * DMODEL * 2);
  bf16*  wtv  = (bf16*) take((size_t)VPAD * DMODEL * 2);

  const int EB = 256;
  k_embed<<<(SEQ * DMODEL + EB - 1) / EB, EB, 0, stream>>>(ids, wte, wpe, x);

  for (int i = 0; i < NL; ++i) {
    // LN1 -> h (bf16)
    k_layernorm<<<SEQ, 256, 0, stream>>>(x, ln1g + i * DMODEL, ln1b + i * DMODEL, h);
    // QKV
    {
      int N = 3 * DMODEL, K = DMODEL;
      k_convT<<<((size_t)N * K + EB - 1) / EB, EB, 0, stream>>>(
          attw + (size_t)i * K * N, wtq, K, N);
      dim3 g(N / 128, SEQ / 64);
      k_gemm_qkv<<<g, 128, 0, stream>>>(h, wtq, attb + (size_t)i * N, qb, kb, vtb);
    }
    // attention
    {
      dim3 gs(SEQ / 16, SEQ / 64, NH);
      k_scores<<<gs, 32, 0, stream>>>(qb, kb, sc);
      k_softmax<<<NH * SEQ, 256, 0, stream>>>(sc, pbuf);
      dim3 go(HD / 32, SEQ / 64, NH);
      k_attnout<<<go, 32, 0, stream>>>(pbuf, vtb, ao);
    }
    // attn output projection + residual
    {
      int N = DMODEL, K = DMODEL;
      k_convT<<<((size_t)N * K + EB - 1) / EB, EB, 0, stream>>>(
          pw + (size_t)i * K * N, wtp, K, N);
      dim3 g(N / 128, SEQ / 64);
      k_gemm<0><<<g, 128, 0, stream>>>(ao, wtp, pb + (size_t)i * N, x, x, nullptr,
                                       N, K, DMODEL, N);
    }
    // LN2 -> h
    k_layernorm<<<SEQ, 256, 0, stream>>>(x, ln2g + i * DMODEL, ln2b + i * DMODEL, h);
    // FC + GELU -> a4 (bf16)
    {
      int N = 4 * DMODEL, K = DMODEL;
      k_convT<<<((size_t)N * K + EB - 1) / EB, EB, 0, stream>>>(
          fcw + (size_t)i * K * N, wtf, K, N);
      dim3 g(N / 128, SEQ / 64);
      k_gemm<1><<<g, 128, 0, stream>>>(h, wtf, fcb + (size_t)i * N, nullptr,
                                       nullptr, a4, N, K, N, N);
    }
    // MLP projection + residual
    {
      int N = DMODEL, K = 4 * DMODEL;
      k_convT<<<((size_t)N * K + EB - 1) / EB, EB, 0, stream>>>(
          prw + (size_t)i * K * N, wtr, K, N);
      dim3 g(N / 128, SEQ / 64);
      k_gemm<0><<<g, 128, 0, stream>>>(a4, wtr, prb + (size_t)i * N, x, x, nullptr,
                                       N, K, DMODEL, N);
    }
  }

  // final LN -> h, then logits = h @ wte^T
  k_layernorm<<<SEQ, 256, 0, stream>>>(x, lnfg, lnfb, h);
  {
    size_t total = (size_t)VPAD * DMODEL, valid = (size_t)VOCAB * DMODEL;
    k_convPad<<<(total + EB - 1) / EB, EB, 0, stream>>>(wte, wtv, total, valid);
    int N = VPAD, K = DMODEL;
    dim3 g((N / 32 + 3) / 4, SEQ / 64);
    k_gemm<2><<<g, 128, 0, stream>>>(h, wtv, nullptr, nullptr, logits, nullptr,
                                     N, K, VOCAB, VOCAB);
  }
}
